// OutputBlock_63531156242865
// MI455X (gfx1250) — matrix operand order repaired and checked by reading, not compiled, not run on hardware
//
#include <hip/hip_runtime.h>
#include <hip/hip_bf16.h>

typedef __attribute__((ext_vector_type(16))) __bf16 v16bf;
typedef __attribute__((ext_vector_type(8)))  float  v8f;
typedef __attribute__((ext_vector_type(4)))  unsigned int u32x4;
typedef __attribute__((ext_vector_type(8)))  int  i32x8;
typedef __attribute__((ext_vector_type(4)))  int  i32x4;

#define N_EDGES_C     1000000
#define N_PARTICLES_C 50000
#define EMBED_C       128
#define OUT_EMBED_C   256

// packed bf16 transposed weights in workspace: [layer][n][k], k contiguous
#define WPACK_L0_OFF  0                        // 256*128
#define WPACK_L1_OFF  (256*128)                // 256*256
#define WPACK_L2_OFF  (256*128 + 256*256)
#define WPACK_L3_OFF  (256*128 + 2*256*256)
#define WPACK_TOTAL   (256*128 + 3*256*256)    // elements (bf16)

// ---------------------------------------------------------------------------
// Kernel A: zero the segment-sum accumulator in workspace (re-run every call).
// ---------------------------------------------------------------------------
__global__ void zero_summed_kernel(float4* __restrict__ p, int n4) {
    int i = blockIdx.x * blockDim.x + threadIdx.x;
    if (i < n4) p[i] = make_float4(0.f, 0.f, 0.f, 0.f);
}

// ---------------------------------------------------------------------------
// Kernel A2: pack all MLP weights to bf16, transposed to [n][k] (k contiguous)
// so the TDM can stream 32-k slabs into LDS as a raw 2-D tile copy.
// ---------------------------------------------------------------------------
__global__ void pack_weights_kernel(const float* __restrict__ W_up,
                                    const float* __restrict__ W1,
                                    const float* __restrict__ W2,
                                    const float* __restrict__ W3,
                                    __bf16* __restrict__ out) {
    const int gid = blockIdx.x * blockDim.x + threadIdx.x;
    if (gid >= WPACK_TOTAL) return;
    int L, off, Kdim;
    if (gid < WPACK_L1_OFF) { L = 0; off = WPACK_L0_OFF; Kdim = 128; }
    else {
        int r = gid - WPACK_L1_OFF;
        L = 1 + r / (256 * 256);
        off = WPACK_L1_OFF + (L - 1) * 256 * 256;
        Kdim = 256;
    }
    const int loc = gid - off;
    const int n = loc / Kdim, k = loc - n * Kdim;
    const float* W = (L == 0) ? W_up : (L == 1) ? W1 : (L == 2) ? W2 : W3;
    out[gid] = (__bf16)W[(size_t)k * OUT_EMBED_C + n];   // transpose: W[k][n] -> out[n][k]
}

// ---------------------------------------------------------------------------
// Kernel B: per-edge  (rbf @ W_rbf) * messages  -> atomic scatter into summed.
// One wave per edge per iteration; lane l owns embed dims [4l, 4l+3].
// W_rbf (16x128) lives in registers: 16 float4 per lane.
// ---------------------------------------------------------------------------
__global__ void edge_gate_scatter_kernel(const float* __restrict__ messages,
                                         const float* __restrict__ rbf,
                                         const int*   __restrict__ idx_i,
                                         const float* __restrict__ W_rbf,
                                         float*       __restrict__ summed,
                                         int n_edges) {
    const int lane = threadIdx.x & 31;
    const int gwave = (blockIdx.x * blockDim.x + threadIdx.x) >> 5;
    const int nwave = (gridDim.x * blockDim.x) >> 5;
    const int d0 = lane * 4;

    float4 w4[16];
#pragma unroll
    for (int r = 0; r < 16; ++r)
        w4[r] = *(const float4*)(W_rbf + r * EMBED_C + d0);

    for (int e = gwave; e < n_edges; e += nwave) {
        const float4* rb = (const float4*)(rbf + (size_t)e * 16);
        float4 r0 = rb[0], r1 = rb[1], r2 = rb[2], r3 = rb[3];
        float rc[16] = {r0.x, r0.y, r0.z, r0.w, r1.x, r1.y, r1.z, r1.w,
                        r2.x, r2.y, r2.z, r2.w, r3.x, r3.y, r3.z, r3.w};
        float t0 = 0.f, t1 = 0.f, t2 = 0.f, t3 = 0.f;
#pragma unroll
        for (int r = 0; r < 16; ++r) {
            const float c = rc[r];
            t0 = fmaf(c, w4[r].x, t0);
            t1 = fmaf(c, w4[r].y, t1);
            t2 = fmaf(c, w4[r].z, t2);
            t3 = fmaf(c, w4[r].w, t3);
        }
        const float4 m = *(const float4*)(messages + (size_t)e * EMBED_C + d0);
        float* dst = summed + (size_t)idx_i[e] * EMBED_C + d0;
        atomicAdd(dst + 0, m.x * t0);
        atomicAdd(dst + 1, m.y * t1);
        atomicAdd(dst + 2, m.z * t2);
        atomicAdd(dst + 3, m.w * t3);
    }
}

// ---------------------------------------------------------------------------
// Kernel C: fused MLP with bf16 WMMA + TDM-double-buffered weight slabs.
// ---------------------------------------------------------------------------
#define MB   64           // rows per block
#define HP   264          // padded LDS row length for h (bf16)
#define KP   34           // padded K length for weight slab rows (bf16)

#if defined(__has_builtin)
#if __has_builtin(__builtin_amdgcn_tensor_load_to_lds)
#define HAVE_TDM 1
#endif
#endif
#ifndef HAVE_TDM
#define HAVE_TDM 0
#endif

#if HAVE_TDM
// Issue one TDM 2-D tile copy: 32 k-columns x 256 n-rows of bf16 from the
// packed weight buffer into LDS, with HW padding of 1 dword after every
// 16 dwords (32 bf16) -> LDS row stride of 34 bf16 (bank-decorrelated).
__device__ __forceinline__ void tdm_load_w_chunk(const __bf16* gbase, int Kdim,
                                                 int kbase, unsigned int lds_off) {
    const unsigned long long ga =
        (unsigned long long)(uintptr_t)gbase + (unsigned long long)kbase * 2ull;
    u32x4 g0;
    g0[0] = 1u;                                              // count=1 (valid D#)
    g0[1] = lds_off;                                         // lds_addr (bytes)
    g0[2] = (unsigned int)ga;                                // global_addr[31:0]
    g0[3] = ((unsigned int)(ga >> 32) & 0x01FFFFFFu)         // global_addr[56:32]
          | (2u << 30);                                      // type=2 ("image")
    i32x8 g1;
    g1[0] = (int)((1u << 16)      // data_size = 2 bytes
                | (1u << 20)      // pad_enable
                | (3u << 22));    // pad_interval = 16 dwords (one 32-bf16 row)
                                  // pad_amount = 0 -> 1 dword pad
    g1[1] = (int)(((unsigned int)Kdim & 0xFFFFu) << 16);     // tensor_dim0[15:0]
    g1[2] = (int)(256u << 16);                               // tensor_dim1 = 256
    g1[3] = (int)(32u << 16);                                // tile_dim0 = 32
    g1[4] = (int)256;                                        // tile_dim1 = 256
    g1[5] = (int)Kdim;                                       // tensor_dim0_stride
    g1[6] = 0; g1[7] = 0;
    i32x4 z4 = {0, 0, 0, 0};
#if __clang_major__ >= 23
    i32x8 z8 = {0, 0, 0, 0, 0, 0, 0, 0};
    __builtin_amdgcn_tensor_load_to_lds(g0, g1, z4, z4, z8, 0);
#else
    __builtin_amdgcn_tensor_load_to_lds(g0, g1, z4, z4, 0);
#endif
}
#endif

__global__ void mlp_wmma_kernel(const float* __restrict__ summed,
                                const __bf16* __restrict__ wpack,
                                const float* __restrict__ b1,
                                const float* __restrict__ b2,
                                const float* __restrict__ b3,
                                const float* __restrict__ W_final,
                                float* __restrict__ out,
                                int nrows) {
    __shared__ __bf16 hbuf[2][MB][HP];       // ping-pong activations (bf16)
    __shared__ __bf16 wT[2][OUT_EMBED_C][KP];// double-buffered weight slabs

    const int tid  = threadIdx.x;
    const int lane = tid & 31;
    const int wave = tid >> 5;            // 0..7
    const int half = lane >> 4;           // 0 or 1
    const int lan16 = lane & 15;
    const int m0 = (wave >> 1) * 16;      // tile row base for this wave
    const int nbase = (wave & 1) * 128;   // this wave covers cols [nbase, nbase+128)
    const int row0 = blockIdx.x * MB;

    // ---- load summed (fp32) -> hbuf[0] as bf16, rows guarded ----
    for (int i = tid; i < MB * EMBED_C; i += blockDim.x) {
        const int r = i >> 7, k = i & 127;
        const int row = row0 + r;
        const float v = (row < nrows) ? summed[(size_t)row * EMBED_C + k] : 0.f;
        hbuf[0][r][k] = (__bf16)v;
    }
    __syncthreads();

    const __bf16* wl[4] = {wpack + WPACK_L0_OFF, wpack + WPACK_L1_OFF,
                           wpack + WPACK_L2_OFF, wpack + WPACK_L3_OFF};
    const int   kdims[4] = {EMBED_C, OUT_EMBED_C, OUT_EMBED_C, OUT_EMBED_C};
    const float* biases[4] = {b1, b1, b2, b3};  // biases[0] unused (no bias on W_up)

    int cur = 0;
    int buf = 0;

#if HAVE_TDM
    // preload layer-0 chunk 0 (single issuing wave; TENSORcnt tracks it)
    if (wave == 0)
        tdm_load_w_chunk(wl[0], kdims[0], 0, (unsigned int)(uintptr_t)&wT[0][0][0]);
#endif

    for (int L = 0; L < 4; ++L) {
        const int Kdim = kdims[L];
        const int nk = Kdim >> 5;

        v8f c[8];
#pragma unroll
        for (int t = 0; t < 8; ++t)
            c[t] = (v8f){0.f, 0.f, 0.f, 0.f, 0.f, 0.f, 0.f, 0.f};

        for (int kc = 0; kc < nk; ++kc) {
            const int kbase = kc << 5;
#if HAVE_TDM
            if (wave == 0) __builtin_amdgcn_s_wait_tensorcnt(0); // chunk kc landed
            __syncthreads();   // publish slab; proves wT[buf^1] readers are done
            if (wave == 0) {   // overlap next DMA with this chunk's WMMAs
                if (kc + 1 < nk)
                    tdm_load_w_chunk(wl[L], Kdim, (kc + 1) << 5,
                                     (unsigned int)(uintptr_t)&wT[buf ^ 1][0][0]);
                else if (L < 3)
                    tdm_load_w_chunk(wl[L + 1], kdims[L + 1], 0,
                                     (unsigned int)(uintptr_t)&wT[buf ^ 1][0][0]);
            }
#else
            // fallback: manual cooperative staging from packed bf16 weights
            __syncthreads();
            for (int i = tid; i < 32 * OUT_EMBED_C; i += blockDim.x) {
                const int n = i >> 5, kk = i & 31;
                wT[buf][n][kk] = wl[L][(size_t)n * Kdim + kbase + kk];
            }
            __syncthreads();
#endif
            // A fragment: 16x32 bf16 tile of hbuf[cur], rows m0..m0+15
            v16bf a;
            {
                const __bf16* hr = &hbuf[cur][m0 + lan16][kbase];
#pragma unroll
                for (int v = 0; v < 8; ++v) {
                    const int koff = ((v < 4) ? (2 * v) : (16 + 2 * (v - 4))) + 8 * half;
                    a[2 * v]     = hr[koff];
                    a[2 * v + 1] = hr[koff + 1];
                }
            }
#pragma unroll
            for (int nt = 0; nt < 8; ++nt) {
                const int n0 = nbase + nt * 16;
                v16bf bfrag;
                const __bf16* wc = &wT[buf][n0 + lan16][0];
#pragma unroll
                for (int v = 0; v < 8; ++v) {
                    const int kk = ((v < 4) ? (2 * v) : (16 + 2 * (v - 4))) + 8 * half;
                    bfrag[2 * v]     = wc[kk];
                    bfrag[2 * v + 1] = wc[kk + 1];
                }
                c[nt] = __builtin_amdgcn_wmma_f32_16x16x32_bf16(
                    false, a, false, bfrag, (short)0, c[nt], false, false);
            }
            buf ^= 1;
        }

        // ---- bias + swish (layers 1..3), write bf16 to other buffer ----
        const int nxt = cur ^ 1;
        const float* __restrict__ bias = biases[L];
#pragma unroll
        for (int nt = 0; nt < 8; ++nt) {
            const int col = nbase + nt * 16 + lan16;
#pragma unroll
            for (int j = 0; j < 8; ++j) {
                const int row = m0 + j + 8 * half;
                float x = c[nt][j];
                if (L) {                        // uniform branch
                    x += bias[col];
                    x = x / (1.f + __expf(-x)); // swish
                }
                hbuf[nxt][row][col] = (__bf16)x;
            }
        }
        __syncthreads();
        cur = nxt;
    }

    // ---- final 256 -> 1 projection ----
    if (tid < MB) {
        const int row = row0 + tid;
        if (row < nrows) {
            float s = 0.f;
#pragma unroll 8
            for (int k = 0; k < OUT_EMBED_C; ++k)
                s = fmaf((float)hbuf[cur][tid][k], W_final[k], s);
            out[row] = s;
        }
    }
}

// ---------------------------------------------------------------------------
extern "C" void kernel_launch(void* const* d_in, const int* in_sizes, int n_in,
                              void* d_out, int out_size, void* d_ws, size_t ws_size,
                              hipStream_t stream) {
    (void)in_sizes; (void)n_in; (void)out_size; (void)ws_size;

    const float* messages = (const float*)d_in[0];
    const float* rbf      = (const float*)d_in[1];
    const int*   idx_i    = (const int*)  d_in[2];
    // d_in[3] = idx_j (unused by reference)
    const float* W_rbf    = (const float*)d_in[4];
    const float* W_up     = (const float*)d_in[5];
    const float* W1       = (const float*)d_in[6];
    const float* b1       = (const float*)d_in[7];
    const float* W2       = (const float*)d_in[8];
    const float* b2       = (const float*)d_in[9];
    const float* W3       = (const float*)d_in[10];
    const float* b3       = (const float*)d_in[11];
    const float* W_final  = (const float*)d_in[12];

    float*  summed = (float*)d_ws;                       // 50000*128 f32 = 25.6 MB
    __bf16* wpack  = (__bf16*)((char*)d_ws +
                     (size_t)N_PARTICLES_C * EMBED_C * sizeof(float)); // +0.46 MB
    float*  outp   = (float*)d_out;                      // 50000 floats

    // A: zero accumulator
    {
        const int n4 = (N_PARTICLES_C * EMBED_C) / 4;
        zero_summed_kernel<<<(n4 + 255) / 256, 256, 0, stream>>>((float4*)summed, n4);
    }
    // A2: pack weights to bf16 transposed (one-shot, ~1 MB traffic)
    pack_weights_kernel<<<(WPACK_TOTAL + 255) / 256, 256, 0, stream>>>(
        W_up, W1, W2, W3, wpack);
    // B: edge stream + gate + scatter
    edge_gate_scatter_kernel<<<2048, 256, 0, stream>>>(
        messages, rbf, idx_i, W_rbf, summed, N_EDGES_C);
    // C: fused WMMA MLP with TDM weight streaming
    mlp_wmma_kernel<<<(N_PARTICLES_C + MB - 1) / MB, 256, 0, stream>>>(
        summed, wpack, b1, b2, b3, W_final, outp, N_PARTICLES_C);
}
